// MultiHeadAttention_68564857913896
// MI455X (gfx1250) — matrix operand order repaired and checked by reading, not compiled, not run on hardware
//
#include <hip/hip_runtime.h>

// ---- problem constants (match reference) ----
#define S_LEN   2048
#define D_MODEL 1024
#define NHEAD   16
#define DK      64
#define NEGV    (-1.0e9f)

typedef __attribute__((ext_vector_type(16))) __bf16 v16bf;
typedef __attribute__((ext_vector_type(8)))  __bf16 v8bf;
typedef __attribute__((ext_vector_type(8)))  float  v8f;

__device__ __forceinline__ v8f wmma_bf16(v16bf a, v16bf b, v8f c) {
  // D = A(16x32 bf16) x B(32x16 bf16) + C(16x16 f32)
  return __builtin_amdgcn_wmma_f32_16x16x32_bf16(false, a, false, b, (short)0, c,
                                                 false, false);
}

__device__ __forceinline__ v8f v8f_zero() {
  return (v8f){0.f, 0.f, 0.f, 0.f, 0.f, 0.f, 0.f, 0.f};
}

// ---- DPP16 butterfly reductions across each 16-lane row (wave32) ----
// quad_perm(1,0,3,2)=0xB1, quad_perm(2,3,0,1)=0x4E,
// row_half_mirror=0x141, row_mirror=0x140.
template <int CTRL>
__device__ __forceinline__ float dpp_bfly(float x) {
  int xi = __float_as_int(x);
  int t  = __builtin_amdgcn_update_dpp(xi, xi, CTRL, 0xF, 0xF, true);
  return __int_as_float(t);
}
__device__ __forceinline__ float red_max16(float x) {
  x = fmaxf(x, dpp_bfly<0xB1>(x));
  x = fmaxf(x, dpp_bfly<0x4E>(x));
  x = fmaxf(x, dpp_bfly<0x141>(x));
  x = fmaxf(x, dpp_bfly<0x140>(x));
  return x;
}
__device__ __forceinline__ float red_sum16(float x) {
  x += dpp_bfly<0xB1>(x);
  x += dpp_bfly<0x4E>(x);
  x += dpp_bfly<0x141>(x);
  x += dpp_bfly<0x140>(x);
  return x;
}

// ---------------- fragment loaders ----------------
// A-matrix 16x32 bf16 layout (ISA 7.12.2): lanes 0-15 hold K=[0..7],[16..23]
// of row M=lane; lanes 16-31 hold K=[8..15],[24..31] of the same rows.
__device__ __forceinline__ v16bf load_a_f32(const float* A, int lda, int mBase,
                                            int k, int lane) {
  int row  = mBase + (lane & 15);
  int koff = k + ((lane >> 4) << 3);
  const float* p = A + (size_t)row * lda + koff;
  v16bf a;
#pragma unroll
  for (int i = 0; i < 8; ++i) { a[i] = (__bf16)p[i]; a[i + 8] = (__bf16)p[i + 16]; }
  return a;
}

__device__ __forceinline__ v16bf load_a_bf16(const __bf16* A, int lda, int mBase,
                                             int k, int lane) {
  int row  = mBase + (lane & 15);
  int koff = k + ((lane >> 4) << 3);
  const __bf16* p = A + (size_t)row * lda + koff;
  v8bf c0 = *(const v8bf*)p;
  v8bf c1 = *(const v8bf*)(p + 16);
  v16bf a;
#pragma unroll
  for (int i = 0; i < 8; ++i) { a[i] = c0[i]; a[i + 8] = c1[i]; }
  return a;
}

// B-matrix 32x16 bf16: lane n (0-15) holds K=[0..7],[16..23] of column nBase+n,
// lanes 16-31 hold K=[8..15],[24..31].  A "column of B" is a row of W / K / V^T,
// i.e. two contiguous 16B runs per lane (koff, koff+16).
__device__ __forceinline__ v16bf load_b_bf16(const __bf16* Bm, int ldb, int nBase,
                                             int k, int lane) {
  int n    = nBase + (lane & 15);
  int koff = k + ((lane >> 4) << 3);
  const __bf16* p = Bm + (size_t)n * ldb + koff;
  v8bf c0 = *(const v8bf*)p;
  v8bf c1 = *(const v8bf*)(p + 16);
  v16bf b;
#pragma unroll
  for (int i = 0; i < 8; ++i) { b[i] = c0[i]; b[i + 8] = c1[i]; }
  return b;
}

// ---------------- weight pre-cast (f32 -> bf16, done once) ----------------
__global__ __launch_bounds__(256) void cast_f32_bf16(const float* __restrict__ in,
                                                     void* __restrict__ outv, int n) {
  __bf16* out = (__bf16*)outv;
  int i = (blockIdx.x * 256 + threadIdx.x) * 4;
  if (i < n) {
    float4 f = *(const float4*)(in + i);
    out[i + 0] = (__bf16)f.x;
    out[i + 1] = (__bf16)f.y;
    out[i + 2] = (__bf16)f.z;
    out[i + 3] = (__bf16)f.w;
  }
}

// ---------------- GEMM: C = X @ W^T ----------------
// W is pre-cast bf16 [N,K]; its 128x32 tile is staged to LDS per K-step via
// CDNA5 async global->LDS copies (ASYNCcnt) and shared by the 4 waves.
// MODE 0: f32 out, row-major [M,N]
// MODE 1: bf16 out, head-split [B,H,S,dk]            (Q / K)
// MODE 2: bf16 out, head-split transposed [B,H,dk,S] (V^T)
// AKIND 0: X is f32 ; AKIND 1: X is bf16
template <int AKIND, int MODE>
__global__ __launch_bounds__(128) void gemm_xwT_wmma(
    const void* __restrict__ Xv, const void* __restrict__ Wv,
    float* __restrict__ outF, void* __restrict__ outBv, int M, int Kdim, int N,
    float oscale) {
  __shared__ __align__(64) __bf16 bTile[128][32];  // 8KB: W rows for this block

  const int tid    = threadIdx.x;
  const int lane   = tid & 31;
  const int wave   = tid >> 5;                      // 4 waves: 2x2 of 64x64
  const int mBase0 = blockIdx.y * 128 + (wave >> 1) * 64;
  const int nBase0 = blockIdx.x * 128 + (wave & 1) * 64;
  const __bf16* Wb = (const __bf16*)Wv;

  const int colg = blockIdx.x * 128 + tid;          // W row owned by this thread
  const unsigned ldsBase = (unsigned)(uintptr_t)&bTile[tid][0];  // LDS offset

  v8f acc[4][4];
#pragma unroll
  for (int i = 0; i < 4; ++i)
#pragma unroll
    for (int j = 0; j < 4; ++j) acc[i][j] = v8f_zero();

  for (int k = 0; k < Kdim; k += 32) {
    __syncthreads();  // previous tile fully consumed
    // async copy W tile: each thread copies one 64B row piece as 4 x b128
#pragma unroll
    for (int q = 0; q < 4; ++q) {
      unsigned long long ga =
          (unsigned long long)(uintptr_t)(Wb + (size_t)colg * Kdim + k + q * 8);
      unsigned la = ldsBase + q * 16;
      asm volatile("global_load_async_to_lds_b128 %0, %1, off"
                   :: "v"(la), "v"(ga) : "memory");
    }
    // A fragments from global while the async copy is in flight
    v16bf af[4];
#pragma unroll
    for (int i = 0; i < 4; ++i) {
      if constexpr (AKIND == 0)
        af[i] = load_a_f32((const float*)Xv, Kdim, mBase0 + 16 * i, k, lane);
      else
        af[i] = load_a_bf16((const __bf16*)Xv, Kdim, mBase0 + 16 * i, k, lane);
    }
    asm volatile("s_wait_asynccnt 0x0" ::: "memory");
    __syncthreads();  // tile visible to all waves

#pragma unroll
    for (int j = 0; j < 4; ++j) {
      int colL = (wave & 1) * 64 + 16 * j + (lane & 15);
      const __bf16* bp = &bTile[colL][(lane >> 4) << 3];
      v8bf b0 = *(const v8bf*)bp;
      v8bf b1 = *(const v8bf*)(bp + 16);
      v16bf bf;
#pragma unroll
      for (int e = 0; e < 8; ++e) { bf[e] = b0[e]; bf[e + 8] = b1[e]; }
#pragma unroll
      for (int i = 0; i < 4; ++i) acc[i][j] = wmma_bf16(af[i], bf, acc[i][j]);
    }
  }

  // C/D layout: lane 0-15 -> N=lane, rows r; lane 16-31 -> rows 8+r.
#pragma unroll
  for (int i = 0; i < 4; ++i)
#pragma unroll
    for (int j = 0; j < 4; ++j)
#pragma unroll
      for (int r = 0; r < 8; ++r) {
        int m = mBase0 + 16 * i + ((lane >> 4) << 3) + r;
        int n = nBase0 + 16 * j + (lane & 15);
        float val = acc[i][j][r] * oscale;
        if constexpr (MODE == 0) {
          outF[(size_t)m * N + n] = val;
        } else {
          __bf16* ob = (__bf16*)outBv;
          int b = m / S_LEN, s = m - b * S_LEN;
          int h = n >> 6, nd = n & 63;
          if constexpr (MODE == 1)
            ob[(((size_t)(b * NHEAD + h)) * S_LEN + s) * DK + nd] = (__bf16)val;
          else
            ob[(((size_t)(b * NHEAD + h)) * DK + nd) * S_LEN + s] = (__bf16)val;
        }
      }
}

// ---------------- flash attention (causal) ----------------
// 8 waves / block; each wave owns one 16-row Q tile; KV streamed in 32-col
// chunks with online softmax.  1/sqrt(dk) is pre-folded into Q.  Only the
// single diagonal chunk needs mask compares; the main loop is mask-free.
// Row max/sum reductions use DPP16 butterflies (pure VALU, no LDS traffic).
__global__ __launch_bounds__(256) void flash_attn_wmma(
    const void* __restrict__ Qv, const void* __restrict__ Kv,
    const void* __restrict__ Vtv, void* __restrict__ Ov) {
  __shared__ __align__(64) __bf16 plds[8][16][32];  // per-wave P staging

  const int lane  = threadIdx.x & 31;
  const int wave  = threadIdx.x >> 5;
  const int bh    = blockIdx.x >> 4;       // (b*H + h)
  const int qgrp  = blockIdx.x & 15;
  const int mBase = (qgrp * 8 + wave) * 16;
  const int b     = bh / NHEAD;
  const int h     = bh - b * NHEAD;
  const int hi    = lane >> 4;
  const int ln    = lane & 15;

  const __bf16* Qh = (const __bf16*)Qv + (size_t)bh * S_LEN * DK;
  const __bf16* Kh = (const __bf16*)Kv + (size_t)bh * S_LEN * DK;
  const __bf16* Vh = (const __bf16*)Vtv + (size_t)bh * DK * S_LEN;

  // Q tile (16 x 64) as two A-fragments, loaded once (already scaled)
  v16bf aQ0 = load_a_bf16(Qh, DK, mBase, 0, lane);
  v16bf aQ1 = load_a_bf16(Qh, DK, mBase, 32, lane);

  v8f o[4];
#pragma unroll
  for (int t = 0; t < 4; ++t) o[t] = v8f_zero();
  float rowM[8], rowS[8];
#pragma unroll
  for (int r = 0; r < 8; ++r) { rowM[r] = -3.0e38f; rowS[r] = 0.f; }

  auto chunk = [&](int jb, bool masked) {
    // scores: two 16x16 tiles (cols jb..+15, jb+16..+31), K-dim 64 = 2 WMMAs
    v8f s0 = v8f_zero();
    v8f s1 = v8f_zero();
    {
      v16bf bK = load_b_bf16(Kh, DK, jb, 0, lane);
      s0 = wmma_bf16(aQ0, bK, s0);
      bK = load_b_bf16(Kh, DK, jb, 32, lane);
      s0 = wmma_bf16(aQ1, bK, s0);
      bK = load_b_bf16(Kh, DK, jb + 16, 0, lane);
      s1 = wmma_bf16(aQ0, bK, s1);
      bK = load_b_bf16(Kh, DK, jb + 16, 32, lane);
      s1 = wmma_bf16(aQ1, bK, s1);
    }
    const int col0 = jb + ln, col1 = col0 + 16;
    float corr[8];
#pragma unroll
    for (int r = 0; r < 8; ++r) {
      int row = mBase + (hi << 3) + r;
      float a = s0[r];
      float c = s1[r];
      if (masked) {                       // folded away in the main loop
        if (col0 > row) a = NEGV;
        if (col1 > row) c = NEGV;
      }
      float mx   = red_max16(fmaxf(a, c));          // DPP butterfly, no LDS
      float newM = fmaxf(rowM[r], mx);
      float cr   = __expf(rowM[r] - newM);
      rowM[r]    = newM;
      float e0 = __expf(a - newM);
      float e1 = __expf(c - newM);
      float ps = red_sum16(e0 + e1);                // DPP butterfly, no LDS
      rowS[r] = rowS[r] * cr + ps;
      corr[r] = cr;
      // C-layout -> row-major bf16 P tile in LDS (for A-frag reload)
      plds[wave][(hi << 3) + r][ln]      = (__bf16)e0;
      plds[wave][(hi << 3) + r][ln + 16] = (__bf16)e1;
    }
    asm volatile("s_wait_dscnt 0" ::: "memory");  // intra-wave LDS RAW
    v16bf aP;
    {
      const __bf16* pr = &plds[wave][ln][hi << 3];
      v8bf c0 = *(const v8bf*)pr;
      v8bf c1 = *(const v8bf*)(pr + 16);
#pragma unroll
      for (int i = 0; i < 8; ++i) { aP[i] = c0[i]; aP[i + 8] = c1[i]; }
    }
    // rescale accumulators, then O += P(16x32) x V(32x64)
#pragma unroll
    for (int t = 0; t < 4; ++t) {
#pragma unroll
      for (int r = 0; r < 8; ++r) o[t][r] *= corr[r];
      v16bf bV = load_b_bf16(Vh + jb, S_LEN, 16 * t, 0, lane);  // V^T rows contiguous
      o[t] = wmma_bf16(aP, bV, o[t]);
    }
  };

  const int jbTail = mBase & ~31;            // only chunk crossing the diagonal
  for (int jb = 0; jb < jbTail; jb += 32) chunk(jb, false);
  chunk(jbTail, true);

  // normalize and write concatenated-head bf16 output [B,S,D]
  float inv[8];
#pragma unroll
  for (int r = 0; r < 8; ++r) inv[r] = 1.0f / rowS[r];
  __bf16* dst = (__bf16*)Ov + (size_t)b * S_LEN * D_MODEL + h * DK;
#pragma unroll
  for (int t = 0; t < 4; ++t)
#pragma unroll
    for (int r = 0; r < 8; ++r) {
      int srow = mBase + (hi << 3) + r;
      dst[(size_t)srow * D_MODEL + 16 * t + ln] = (__bf16)(o[t][r] * inv[r]);
    }
}

// ---------------- host launcher ----------------
extern "C" void kernel_launch(void* const* d_in, const int* in_sizes, int n_in,
                              void* d_out, int out_size, void* d_ws, size_t ws_size,
                              hipStream_t stream) {
  (void)n_in; (void)out_size; (void)ws_size;
  const float* q  = (const float*)d_in[0];
  const float* k  = (const float*)d_in[1];
  const float* v  = (const float*)d_in[2];
  // d_in[3] = mask: causal tril, computed analytically in-kernel
  const float* wq = (const float*)d_in[4];
  const float* wk = (const float*)d_in[5];
  const float* wv = (const float*)d_in[6];
  const float* wo = (const float*)d_in[7];
  float* out = (float*)d_out;

  const int B = in_sizes[0] / (S_LEN * D_MODEL);
  const int M = B * S_LEN;
  const int NW = D_MODEL * D_MODEL;  // weight element count

  char* ws = (char*)d_ws;
  const size_t seg  = (size_t)B * NHEAD * S_LEN * DK * sizeof(__bf16);  // 8 MB @ B=2
  const size_t wseg = (size_t)NW * sizeof(__bf16);                      // 2 MB
  void* Qb  = (void*)(ws);
  void* Kb  = (void*)(ws + seg);
  void* Vtb = (void*)(ws + 2 * seg);
  void* Ab  = (void*)(ws + 3 * seg);            // attention out, bf16 [B,S,D]
  void* Wq  = (void*)(ws + 4 * seg);
  void* Wk  = (void*)(ws + 4 * seg + wseg);
  void* Wv_ = (void*)(ws + 4 * seg + 2 * wseg);
  void* Wo  = (void*)(ws + 4 * seg + 3 * wseg);

  dim3 gc(NW / (256 * 4));
  cast_f32_bf16<<<gc, 256, 0, stream>>>(wq, Wq, NW);
  cast_f32_bf16<<<gc, 256, 0, stream>>>(wk, Wk, NW);
  cast_f32_bf16<<<gc, 256, 0, stream>>>(wv, Wv_, NW);
  cast_f32_bf16<<<gc, 256, 0, stream>>>(wo, Wo, NW);

  const float qscale = 0.125f;  // 1/sqrt(dk), folded into Q projection
  dim3 gg(D_MODEL / 128, M / 128);
  gemm_xwT_wmma<0, 1><<<gg, 128, 0, stream>>>(q, Wq, nullptr, Qb, M, D_MODEL, D_MODEL, qscale);
  gemm_xwT_wmma<0, 1><<<gg, 128, 0, stream>>>(k, Wk, nullptr, Kb, M, D_MODEL, D_MODEL, 1.0f);
  gemm_xwT_wmma<0, 2><<<gg, 128, 0, stream>>>(v, Wv_, nullptr, Vtb, M, D_MODEL, D_MODEL, 1.0f);

  dim3 ga(B * NHEAD * (S_LEN / 16 / 8));
  flash_attn_wmma<<<ga, 256, 0, stream>>>(Qb, Kb, Vtb, Ab);

  gemm_xwT_wmma<1, 0><<<gg, 128, 0, stream>>>(Ab, Wo, out, nullptr, M, D_MODEL, D_MODEL, 1.0f);
}
